// GraphSAGE_area_sobel_77180562309266
// MI455X (gfx1250) — compile-verified, matrix operand order checked by reading, not run on hardware
//
#include <hip/hip_runtime.h>
#include <hip/hip_bf16.h>

// ---------------- problem constants ----------------
#define NN  163840      // N_NODES
#define NE  2621440     // N_EDGES
#define IND 128
#define HID 256
#define OUTD 64
#define BATCH 32768

typedef unsigned short u16;
typedef unsigned int   u32;
typedef __attribute__((ext_vector_type(16))) __bf16 v16bf;
typedef __attribute__((ext_vector_type(2)))  __bf16 v2bf;
typedef __attribute__((ext_vector_type(8)))  float  v8f;
typedef __attribute__((ext_vector_type(8)))  u16    v8u16;
typedef __attribute__((ext_vector_type(16))) u16    v16u16;

union BFCast { v16u16 u; v16bf b; };
union PKCast { v2bf h; u32 w; };

// hardware packed f32 -> bf16 conversion (v_cvt_pk_bf16_f32 on gfx1250)
__device__ __forceinline__ u32 f2bf_pk(float a, float b) {
    PKCast c;
    c.h = (v2bf){(__bf16)a, (__bf16)b};
    return c.w;
}

__device__ __forceinline__ u16 f2bf(float f) {
    unsigned u = __float_as_uint(f);
    unsigned r = u + 0x7FFFu + ((u >> 16) & 1u);   // RNE
    return (u16)(r >> 16);
}

// ---------------- utility kernels ----------------
__global__ void k_fill0(float* __restrict__ p, size_t n) {
    size_t i = (size_t)blockIdx.x * blockDim.x + threadIdx.x;
    size_t stride = (size_t)gridDim.x * blockDim.x;
    for (; i < n; i += stride) p[i] = 0.0f;
}

__global__ void k_degree(const long long* __restrict__ ei, float* __restrict__ deg) {
    size_t e = (size_t)blockIdx.x * blockDim.x + threadIdx.x;
    if (e >= NE) return;
    long long dst = ei[(size_t)NE + e];
    atomicAdd(&deg[dst], 1.0f);
}

__global__ void k_invdeg(float* __restrict__ deg) {
    size_t i = (size_t)blockIdx.x * blockDim.x + threadIdx.x;
    if (i >= NN) return;
    deg[i] = 1.0f / fmaxf(deg[i], 1.0f);
}

// concat [wl | wr] row-major -> bf16 B[n][Ka+Kb]
__global__ void k_cvt_w(const float* __restrict__ wl, const float* __restrict__ wr,
                        int Ka, int Kb, int Nrows, u16* __restrict__ out) {
    int Kt = Ka + Kb;
    size_t total = (size_t)Nrows * Kt;
    size_t i = (size_t)blockIdx.x * blockDim.x + threadIdx.x;
    if (i >= total) return;
    int n = (int)(i / Kt);
    int k = (int)(i % Kt);
    float v = (k < Ka) ? wl[(size_t)n * Ka + k] : wr[(size_t)n * Kb + (k - Ka)];
    out[i] = f2bf(v);
}

// edge aggregation: agg[dst] += x[src], D floats per node, 4 floats per thread
__global__ void k_aggregate(const float* __restrict__ x, const long long* __restrict__ ei,
                            float* __restrict__ agg, int D) {
    int D4 = D >> 2;
    size_t total = (size_t)NE * D4;
    size_t gid = (size_t)blockIdx.x * blockDim.x + threadIdx.x;
    if (gid >= total) return;
    size_t e = gid / D4;
    int c = (int)(gid % D4) << 2;
    long long src = ei[e];
    long long dst = ei[(size_t)NE + e];
    const float4 v = *(const float4*)(x + (size_t)src * D + c);
    float* a = agg + (size_t)dst * D + c;
    atomicAdd(a + 0, v.x);
    atomicAdd(a + 1, v.y);
    atomicAdd(a + 2, v.z);
    atomicAdd(a + 3, v.w);
}

// ---------------- fused SAGE GEMM ----------------
// out[m][n] = act( [agg*invd | xr](m,:) @ B[n][:]^T + bias[n] )
// A tile 64 x 32 staged as bf16 in LDS (fp32->bf16 cvt + inv_deg scale).
// B tile Nout x 32 bf16 staged via GLOBAL_LOAD_ASYNC_TO_LDS (ASYNCcnt).
// Per wave: 2 M-subtiles (mh = wave&1) x NTW N-subtiles (nc = wave>>1).
#define LDA 40   // padded row stride (u16 units), 80B = 16B aligned
template<int TPW>
__global__ __launch_bounds__(256) void k_gemm_sage(
        const float* __restrict__ aggp, const float* __restrict__ invd,
        const float* __restrict__ xr, int Ka, int Kb,
        const u16* __restrict__ wB, const float* __restrict__ bias,
        float* __restrict__ out, int Nout, int relu) {

    constexpr int NTW = TPW / 2;     // N-subtiles per wave
    __shared__ u16 As[64 * LDA];     // 5.0 KB
    __shared__ u16 Bs[256 * LDA];    // 20.0 KB

    const int t    = threadIdx.x;
    const int wv   = t >> 5;
    const int lane = t & 31;
    const int lr   = lane & 15;
    const int lh   = lane >> 4;      // 0 or 1
    const int kb   = lh << 3;        // 0 or 8
    const int mh   = wv & 1;         // which pair of M-subtiles
    const int nc   = wv >> 1;        // N-subtile column group (0..3)
    const int m0   = blockIdx.x * 64;
    const int Ktot = Ka + Kb;
    const int steps = Ktot >> 5;

    v8f acc[TPW];
#pragma unroll
    for (int i = 0; i < TPW; ++i) acc[i] = (v8f){0,0,0,0,0,0,0,0};

    for (int s = 0; s < steps; ++s) {
        const int k0 = s << 5;

        // ---- stage B: Nout rows x 32 K bf16, async global -> LDS ----
        if (t < Nout) {   // wave-uniform (Nout is 256 or 64)
            const u16* gp = wB + (size_t)t * Ktot + k0;        // 64B-aligned
            u32 ldsoff = (u32)(uintptr_t)(&Bs[t * LDA]);       // LDS aperture: addr[31:0]
            asm volatile(
                "global_load_async_to_lds_b128 %0, %1, off\n\t"
                "global_load_async_to_lds_b128 %0, %1, off offset:16\n\t"
                "global_load_async_to_lds_b128 %0, %1, off offset:32\n\t"
                "global_load_async_to_lds_b128 %0, %1, off offset:48"
                :: "v"(ldsoff), "v"(gp) : "memory");
        }

        // ---- stage A: 64 rows x 32 K, fp32 -> bf16 (with inv_deg scaling) ----
        {
            const int row = t >> 2;
            const int kc  = (t & 3) << 3;
            const int grow = m0 + row;
            const int gk   = k0 + kc;
            u32* d = (u32*)&As[row * LDA + kc];   // 16B-aligned
            float v[8];
            if (gk < Ka) {
                const float* sp = aggp + (size_t)grow * Ka + gk;
                const float sc = invd[grow];
#pragma unroll
                for (int j = 0; j < 8; ++j) v[j] = sp[j] * sc;
            } else {
                const float* sp = xr + (size_t)grow * Kb + (gk - Ka);
#pragma unroll
                for (int j = 0; j < 8; ++j) v[j] = sp[j];
            }
#pragma unroll
            for (int j = 0; j < 4; ++j) d[j] = f2bf_pk(v[2 * j], v[2 * j + 1]);
        }

        asm volatile("s_wait_asynccnt 0x0" ::: "memory");
        __syncthreads();

        // ---- A fragments: two M-subtiles, shared across all N-subtiles ----
        BFCast fa[2];
#pragma unroll
        for (int p = 0; p < 2; ++p) {
            const int mt = 2 * mh + p;
            const u16* ap = &As[(mt * 16 + lr) * LDA + kb];
            v8u16 lo = *(const v8u16*)(ap);        // K = kb .. kb+7
            v8u16 hi = *(const v8u16*)(ap + 16);   // K = kb+16 .. kb+23
            fa[p].u = __builtin_shufflevector(lo, hi, 0,1,2,3,4,5,6,7,8,9,10,11,12,13,14,15);
        }

        // ---- for each B fragment: 2 WMMAs (both M-subtiles) ----
#pragma unroll
        for (int j = 0; j < NTW; ++j) {
            const int nt = nc + 4 * j;
            BFCast fb;
            const u16* bp = &Bs[(nt * 16 + lr) * LDA + kb];
            v8u16 lo = *(const v8u16*)(bp);
            v8u16 hi = *(const v8u16*)(bp + 16);
            fb.u = __builtin_shufflevector(lo, hi, 0,1,2,3,4,5,6,7,8,9,10,11,12,13,14,15);
#pragma unroll
            for (int p = 0; p < 2; ++p) {
                acc[2 * j + p] = __builtin_amdgcn_wmma_f32_16x16x32_bf16(
                    false, fa[p].b, false, fb.b, (short)0, acc[2 * j + p], false, false);
            }
        }
        __syncthreads();
    }

    // ---- epilogue: bias + optional ReLU, per documented C/D layout ----
#pragma unroll
    for (int j = 0; j < NTW; ++j) {
        const int nt = nc + 4 * j;
        const int n  = nt * 16 + lr;
        const float b = bias[n];
#pragma unroll
        for (int p = 0; p < 2; ++p) {
            const int mt = 2 * mh + p;
#pragma unroll
            for (int r = 0; r < 8; ++r) {
                const int m = m0 + mt * 16 + lh * 8 + r;
                float v = acc[2 * j + p][r] + b;
                if (relu) v = fmaxf(v, 0.0f);
                out[(size_t)m * Nout + n] = v;
            }
        }
    }
}

// ---------------- host-side orchestration ----------------
extern "C" void kernel_launch(void* const* d_in, const int* in_sizes, int n_in,
                              void* d_out, int out_size, void* d_ws, size_t ws_size,
                              hipStream_t stream) {
    const float*     x    = (const float*)d_in[0];
    const long long* ei   = (const long long*)d_in[1];
    const float*     wl1  = (const float*)d_in[2];
    const float*     wr1  = (const float*)d_in[3];
    const float*     bl1  = (const float*)d_in[4];
    const float*     wl   = (const float*)d_in[5];   // [4][256][256]
    const float*     wr   = (const float*)d_in[6];   // [4][256][256]
    const float*     bl   = (const float*)d_in[7];   // [4][256]
    const float*     fcw  = (const float*)d_in[8];   // [64][1280]
    const float*     fcb  = (const float*)d_in[9];
    float* outp = (float*)d_out;

    // workspace carve-up
    float* deg  = (float*)d_ws;                       // NN
    float* agg  = deg + NN;                           // NN*256
    float* hA   = agg + (size_t)NN * HID;             // NN*256
    float* hB   = hA  + (size_t)NN * HID;             // NN*256
    u16*   wbf1 = (u16*)(hB + (size_t)NN * HID);      // 256*256
    u16*   wbfL = wbf1 + (size_t)HID * (IND + IND);   // 4 * 256*512
    u16*   wbfF = wbfL + (size_t)4 * HID * (HID + HID); // 64*1280

    // degree / inv_deg
    k_fill0<<<2048, 256, 0, stream>>>(deg, (size_t)NN);
    k_degree<<<(NE + 255) / 256, 256, 0, stream>>>(ei, deg);
    k_invdeg<<<(NN + 255) / 256, 256, 0, stream>>>(deg);

    // weight conversion to bf16, [n][k_l | k_r] concat layout
    {
        size_t tot = (size_t)HID * (IND + IND);
        k_cvt_w<<<(unsigned)((tot + 255) / 256), 256, 0, stream>>>(wl1, wr1, IND, IND, HID, wbf1);
        for (int l = 0; l < 4; ++l) {
            size_t tl = (size_t)HID * (HID + HID);
            k_cvt_w<<<(unsigned)((tl + 255) / 256), 256, 0, stream>>>(
                wl + (size_t)l * HID * HID, wr + (size_t)l * HID * HID,
                HID, HID, HID, wbfL + (size_t)l * HID * (HID + HID));
        }
        size_t tf = (size_t)OUTD * (5 * HID);
        k_cvt_w<<<(unsigned)((tf + 255) / 256), 256, 0, stream>>>(fcw, fcw, 5 * HID, 0, OUTD, wbfF);
    }

    // ---- layer 1: in=128 -> 256 ----
    k_fill0<<<8192, 256, 0, stream>>>(agg, (size_t)NN * IND);
    {
        size_t tot = (size_t)NE * (IND / 4);
        k_aggregate<<<(unsigned)(tot / 256), 256, 0, stream>>>(x, ei, agg, IND);
    }
    k_gemm_sage<8><<<NN / 64, 256, 0, stream>>>(agg, deg, x, IND, IND,
                                                wbf1, bl1, hA, HID, 1);

    // ---- layers 2..5: 256 -> 256, ping-pong hA/hB ----
    float* hcur = hA;
    float* hnxt = hB;
    for (int l = 0; l < 4; ++l) {
        k_fill0<<<16384, 256, 0, stream>>>(agg, (size_t)NN * HID);
        size_t tot = (size_t)NE * (HID / 4);
        k_aggregate<<<(unsigned)(tot / 256), 256, 0, stream>>>(hcur, ei, agg, HID);
        k_gemm_sage<8><<<NN / 64, 256, 0, stream>>>(
            agg, deg, hcur, HID, HID,
            wbfL + (size_t)l * HID * (HID + HID), bl + (size_t)l * HID,
            hnxt, HID, 1);
        float* tswap = hcur; hcur = hnxt; hnxt = tswap;
    }

    // ---- final FC: [32768, 1280] @ fc_w.T + fc_b ----
    k_gemm_sage<2><<<BATCH / 64, 256, 0, stream>>>(
        (const float*)nullptr, deg, hcur, 0, 5 * HID,
        wbfF, fcb, outp, OUTD, 0);
}
